// SpatialAttention_80547816669482
// MI455X (gfx1250) — compile-verified, matrix operand order checked by reading, not run on hardware
//
#include <hip/hip_runtime.h>
#include <hip/hip_bf16.h>

typedef __attribute__((ext_vector_type(16))) _Float16 v16h;
typedef __attribute__((ext_vector_type(8)))  _Float16 v8h;
typedef __attribute__((ext_vector_type(4)))  _Float16 v4h;
typedef __attribute__((ext_vector_type(8)))  float    v8f;
typedef __attribute__((ext_vector_type(4)))  float    v4f;
typedef __attribute__((ext_vector_type(4)))  unsigned int v4u;
typedef __attribute__((ext_vector_type(8)))  int      v8i;
typedef __attribute__((ext_vector_type(4)))  int      v4i;

typedef v4f v4f_alias __attribute__((may_alias));
typedef v4h v4h_alias __attribute__((may_alias));

#define B_      2
#define C_      192
#define HEADS_  6
#define HD_     32
#define L_      65536                  // 256*256 pixels
#define SCALE_  0.17677669529663687f   // 32^-0.5
#define MPOS_   961                    // (2*16-1)^2 bias rows

// ---- dynamic LDS layout for the attention kernel (bytes) ----
#define SC_STRIDE_F 260          // score row stride in floats (256 + 4 pad -> conflict-free)
#define SC_STRIDE_H 520          // same row stride in halves
#define Q_OFF    266240          // scores: 256 * 1040 = 266240 bytes
#define K_OFF    282624          // Q f16 256x32 = 16384
#define VT_OFF   299008          // K f16 256x32 = 16384
#define POS_OFF  315392          // Vt f16 32x256 = 16384
#define RINV_OFF 319236          // pos slice 961 f32 = 3844
#define SMEM_BYTES 320260        // + 256 f32 rowsum-inv = 1024

// Build a WMMA 16-bit operand fragment from LDS.
// Per the CDNA5 ISA 16-bit A layout (symmetric assumption for B):
//   lane<16  holds K {0..7, 16..23}, lane>=16 holds K {8..15, 24..31}.
// Caller passes p already offset by (sel*8) halves; second chunk is +16 halves.
static __device__ __forceinline__ v16h ld_frag(const _Float16* p) {
  v8h lo = *(const v8h*)p;
  v8h hi = *(const v8h*)(p + 16);
  v16h r;
#pragma unroll
  for (int t = 0; t < 8; ++t) { r[t] = lo[t]; r[t + 8] = hi[t]; }
  return r;
}

// ---------------- Kernel A: DynamicPosBias MLP -> ws (961 x 6 floats) --------
__global__ __launch_bounds__(256) void pos_mlp_kernel(
    const float* __restrict__ pw,  const float* __restrict__ pb,
    const float* __restrict__ g1,  const float* __restrict__ b1,
    const float* __restrict__ f1,  const float* __restrict__ fb1,
    const float* __restrict__ g2,  const float* __restrict__ b2,
    const float* __restrict__ f2,  const float* __restrict__ fb2,
    const float* __restrict__ g3,  const float* __restrict__ b3,
    const float* __restrict__ f3,  const float* __restrict__ fb3,
    float* __restrict__ pos_out)
{
  __shared__ float sw[5430];
  const int tid = threadIdx.x;
#define STAGE(src, off, cnt) for (int i = tid; i < (cnt); i += 256) sw[(off) + i] = (src)[i];
  STAGE(pw,    0,   96)  STAGE(pb,   96,   48)
  STAGE(g1,  144,   48)  STAGE(b1,  192,   48)
  STAGE(f1,  240, 2304)  STAGE(fb1, 2544,  48)
  STAGE(g2, 2592,   48)  STAGE(b2, 2640,   48)
  STAGE(f2, 2688, 2304)  STAGE(fb2, 4992,  48)
  STAGE(g3, 5040,   48)  STAGE(b3, 5088,   48)
  STAGE(f3, 5136,  288)  STAGE(fb3, 5424,   6)
#undef STAGE
  __syncthreads();

  for (int r = tid; r < MPOS_; r += 256) {
    const float hh = (float)(r / 31) - 15.f;
    const float wv = (float)(r % 31) - 15.f;
    float p[48], t[48];
    for (int c = 0; c < 48; ++c) p[c] = hh * sw[c] + wv * sw[48 + c] + sw[96 + c];

    auto ln_relu = [&](int go, int bo) {
      float m = 0.f;
      for (int k = 0; k < 48; ++k) m += p[k];
      m *= (1.f / 48.f);
      float v = 0.f;
      for (int k = 0; k < 48; ++k) { float d = p[k] - m; v += d * d; }
      v *= (1.f / 48.f);
      const float rs = rsqrtf(v + 1e-5f);
      for (int k = 0; k < 48; ++k)
        t[k] = fmaxf(0.f, (p[k] - m) * rs * sw[go + k] + sw[bo + k]);
    };

    ln_relu(144, 192);
    for (int c = 0; c < 48; ++c) {
      float a = sw[2544 + c];
      for (int k = 0; k < 48; ++k) a += t[k] * sw[240 + k * 48 + c];
      p[c] = a;
    }
    ln_relu(2592, 2640);
    for (int c = 0; c < 48; ++c) {
      float a = sw[4992 + c];
      for (int k = 0; k < 48; ++k) a += t[k] * sw[2688 + k * 48 + c];
      p[c] = a;
    }
    ln_relu(5040, 5088);
    for (int j = 0; j < 6; ++j) {
      float a = sw[5424 + j];
      for (int k = 0; k < 48; ++k) a += t[k] * sw[5136 + k * 6 + j];
      pos_out[r * 6 + j] = a;
    }
  }
}

// ---------------- Kernel B: windowed attention, one block per (window, head) -
__global__ __launch_bounds__(256) void win_attn_kernel(
    const float* __restrict__ qkv, const float* __restrict__ pos,
    float* __restrict__ out)
{
  extern __shared__ char smem[];
  float*    sc   = (float*)smem;                   // scores f32 -> P f16 in place
  _Float16* qs   = (_Float16*)(smem + Q_OFF);      // Q f16 256x32 (pre-scaled)
  _Float16* ks   = (_Float16*)(smem + K_OFF);      // K f16 256x32
  _Float16* vt   = (_Float16*)(smem + VT_OFF);     // V^T f16 32x256
  float*    posl = (float*)(smem + POS_OFF);       // per-head bias slice (961)
  float*    rinv = (float*)(smem + RINV_OFF);      // 1/rowsum (256)

  const int tid  = threadIdx.x;
  const int lane = tid & 31;
  const int wave = tid >> 5;
  const int sel  = lane >> 4;       // 0: K{0..7,16..23}; 1: K{8..15,24..31}
  const int l15  = lane & 15;

  const int win  = blockIdx.x;      // 0..511
  const int head = blockIdx.y;      // 0..5
  const int b    = win >> 8;
  const int wy   = (win >> 4) & 15;
  const int wx   = win & 15;

  // ---- stage this head's bias column into LDS ----
  // 2-D TDM tile: 961 rows x 1 element, row stride 6 floats -> contiguous LDS.
#if __has_builtin(__builtin_amdgcn_tensor_load_to_lds) && __has_builtin(__builtin_amdgcn_s_wait_tensorcnt)
#define USE_TDM_POS 1
  if (wave == 0) {
    const unsigned long long ga = (unsigned long long)(const void*)(pos + head);
    v4u g0;
    g0[0] = 1u;                                    // count=1, user descriptor
    g0[1] = (unsigned)POS_OFF;                     // lds_addr (dynamic LDS starts at 0)
    g0[2] = (unsigned)(ga & 0xffffffffu);          // global_addr[31:0]
    g0[3] = (unsigned)((ga >> 32) & 0x01ffffffu)   // global_addr[56:32]
            | (2u << 30);                          // type=2 ("image")
    v8i g1;
    g1[0] = (int)(2u << 16);                       // data_size=4B; wg_mask=0
    g1[1] = (int)(6u << 16);                       // tensor_dim0=6 (lo16)
    g1[2] = (int)(961u << 16);                     // tensor_dim0 hi=0 | tensor_dim1 lo16=961
    g1[3] = (int)(1u << 16);                       // tensor_dim1 hi=0 | tile_dim0=1
    g1[4] = 961;                                   // tile_dim1=961, tile_dim2=0
    g1[5] = 6;                                     // tensor_dim0_stride=6 (lo32)
    g1[6] = 0;                                     // stride hi | dim1_stride lo (unused, 2D)
    g1[7] = 0;
    v4i gz = {0, 0, 0, 0};
#if __clang_major__ >= 23
    v8i gz8 = {0, 0, 0, 0, 0, 0, 0, 0};
    __builtin_amdgcn_tensor_load_to_lds(g0, g1, gz, gz, gz8, 0);
#else
    __builtin_amdgcn_tensor_load_to_lds(g0, g1, gz, gz, 0);
#endif
  }
#else
#define USE_TDM_POS 0
  for (int i2 = tid; i2 < MPOS_; i2 += 256) posl[i2] = pos[i2 * 6 + head];
#endif

  // ---- load Q/K/V rows (thread n owns window row n), convert to f16 ----
  // Streaming data (touched exactly once grid-wide): non-temporal loads.
  {
    const int n   = tid;
    const int pix = (wy * 16 + (n >> 4)) * 256 + (wx * 16 + (n & 15));
    const size_t TS   = (size_t)B_ * L_ * C_;                        // tensor stride
    const size_t base = (size_t)b * L_ * C_ + (size_t)pix * C_ + head * HD_;
    const float* qp = qkv + base;
    const float* kp = qp + TS;
    const float* vp = kp + TS;
    __builtin_prefetch(kp, 0, 3);   // gfx1250 global_prefetch_b8
    __builtin_prefetch(vp, 0, 3);
#pragma unroll
    for (int dd = 0; dd < HD_; dd += 4) {
      const v4f q4 = __builtin_nontemporal_load((const v4f_alias*)(qp + dd));
      const v4f k4 = __builtin_nontemporal_load((const v4f_alias*)(kp + dd));
      const v4f v4 = __builtin_nontemporal_load((const v4f_alias*)(vp + dd));
      v4h qh = {(_Float16)(q4.x * SCALE_), (_Float16)(q4.y * SCALE_),
                (_Float16)(q4.z * SCALE_), (_Float16)(q4.w * SCALE_)};
      v4h kh = {(_Float16)k4.x, (_Float16)k4.y, (_Float16)k4.z, (_Float16)k4.w};
      *(v4h*)(qs + n * 32 + dd) = qh;
      *(v4h*)(ks + n * 32 + dd) = kh;
      vt[(dd + 0) * 256 + n] = (_Float16)v4.x;
      vt[(dd + 1) * 256 + n] = (_Float16)v4.y;
      vt[(dd + 2) * 256 + n] = (_Float16)v4.z;
      vt[(dd + 3) * 256 + n] = (_Float16)v4.w;
    }
  }
#if USE_TDM_POS
  if (wave == 0) __builtin_amdgcn_s_wait_tensorcnt(0);
#endif
  __syncthreads();

  // ---- S = Q.K^T + bias: wave w owns query tile-rows {2w, 2w+1}.
  //      Bias is fed in as the WMMA C-matrix (8 consecutive posl floats/lane).
#pragma unroll
  for (int ii = 0; ii < 2; ++ii) {
    const int i = wave * 2 + ii;
    const v16h A = ld_frag(qs + (i * 16 + l15) * 32 + sel * 8);
    for (int j = 0; j < 16; ++j) {
      const v16h Bf = ld_frag(ks + (j * 16 + l15) * 32 + sel * 8);
      const int bb = (i - j + 15) * 31 + 15 - l15 + sel * 8;  // consecutive over r
      v8f c;
#pragma unroll
      for (int r = 0; r < 8; ++r) c[r] = posl[bb + r];
      const v8f d = __builtin_amdgcn_wmma_f32_16x16x32_f16(
          false, A, false, Bf, (short)0, c, false, false);
      float* srow = sc + (i * 16 + sel * 8) * SC_STRIDE_F + j * 16 + l15;
#pragma unroll
      for (int r = 0; r < 8; ++r) srow[r * SC_STRIDE_F] = d[r];
    }
  }
  __syncthreads();

  // ---- softmax: thread n owns row n; exp written back in place as f16 ----
  {
    float* row = sc + tid * SC_STRIDE_F;
    float mx = -3.0e38f;
    for (int m2 = 0; m2 < 256; m2 += 4) {
      const v4f x = *(const v4f_alias*)(row + m2);
      mx = fmaxf(mx, fmaxf(fmaxf(x.x, x.y), fmaxf(x.z, x.w)));
    }
    float s = 0.f;
    _Float16* prow = (_Float16*)row;
    for (int m2 = 0; m2 < 256; m2 += 4) {
      const v4f x = *(const v4f_alias*)(row + m2);     // read f32 ahead of f16 write
      const float e0 = __expf(x.x - mx), e1 = __expf(x.y - mx);
      const float e2 = __expf(x.z - mx), e3 = __expf(x.w - mx);
      s += (e0 + e1) + (e2 + e3);
      v4h h4 = {(_Float16)e0, (_Float16)e1, (_Float16)e2, (_Float16)e3};
      *(v4h_alias*)(prow + m2) = h4;                    // front-packed, already-read bytes
    }
    rinv[tid] = 1.0f / s;
  }
  __syncthreads();

  // ---- O = P.V (K=256 in 8 WMMA steps); fold 1/rowsum into the NT store ----
#pragma unroll
  for (int ii = 0; ii < 2; ++ii) {
    const int i = wave * 2 + ii;
    const _Float16* prow =
        (const _Float16*)smem + (size_t)(i * 16 + l15) * SC_STRIDE_H + sel * 8;
#pragma unroll
    for (int dblk = 0; dblk < 2; ++dblk) {
      const _Float16* vrow = vt + (dblk * 16 + l15) * 256 + sel * 8;
      v8f acc = {0.f, 0.f, 0.f, 0.f, 0.f, 0.f, 0.f, 0.f};
#pragma unroll
      for (int kc = 0; kc < 8; ++kc) {
        const v16h A  = ld_frag(prow + kc * 32);
        const v16h Bf = ld_frag(vrow + kc * 32);
        acc = __builtin_amdgcn_wmma_f32_16x16x32_f16(
            false, A, false, Bf, (short)0, acc, false, false);
      }
#pragma unroll
      for (int r = 0; r < 8; ++r) {
        const int Mr   = r + sel * 8;
        const int nrow = i * 16 + Mr;
        const float val = acc[r] * rinv[nrow];
        const int pix = (wy * 16 + i) * 256 + (wx * 16 + Mr);
        __builtin_nontemporal_store(
            val, out + ((size_t)b * L_ + pix) * C_ + head * HD_ + dblk * 16 + l15);
      }
    }
  }
}

// ---------------------------------------------------------------------------
extern "C" void kernel_launch(void* const* d_in, const int* in_sizes, int n_in,
                              void* d_out, int out_size, void* d_ws, size_t ws_size,
                              hipStream_t stream) {
  const float* qkv = (const float*)d_in[0];
  const float* pw  = (const float*)d_in[1];
  const float* pb  = (const float*)d_in[2];
  const float* g1  = (const float*)d_in[3];
  const float* b1  = (const float*)d_in[4];
  const float* f1  = (const float*)d_in[5];
  const float* fb1 = (const float*)d_in[6];
  const float* g2  = (const float*)d_in[7];
  const float* b2  = (const float*)d_in[8];
  const float* f2  = (const float*)d_in[9];
  const float* fb2 = (const float*)d_in[10];
  const float* g3  = (const float*)d_in[11];
  const float* b3  = (const float*)d_in[12];
  const float* f3  = (const float*)d_in[13];
  const float* fb3 = (const float*)d_in[14];
  float* out    = (float*)d_out;
  float* pos_ws = (float*)d_ws;   // 961*6 floats of scratch

  pos_mlp_kernel<<<1, 256, 0, stream>>>(pw, pb, g1, b1, f1, fb1, g2, b2,
                                        f2, fb2, g3, b3, f3, fb3, pos_ws);

  (void)hipFuncSetAttribute((const void*)win_attn_kernel,
                            hipFuncAttributeMaxDynamicSharedMemorySize, SMEM_BYTES);
  win_attn_kernel<<<dim3(512, HEADS_), 256, SMEM_BYTES, stream>>>(qkv, pos_ws, out);
}
